// InitialNetwork_10857677324344
// MI455X (gfx1250) — matrix operand order repaired and checked by reading, not compiled, not run on hardware
//
#include <hip/hip_runtime.h>
#include <hip/hip_bf16.h>

typedef __bf16 v16bf __attribute__((ext_vector_type(16)));
typedef __bf16 v8bf  __attribute__((ext_vector_type(8)));
typedef __bf16 v2bf  __attribute__((ext_vector_type(2)));
typedef float  v8f   __attribute__((ext_vector_type(8)));

#define B_    256
#define T_    512
#define OBS_  256
#define H_    128
#define N4_   512   // 4*H
#define LAT_  16

__device__ __forceinline__ float sigmoidf_(float x) {
    return 1.0f / (1.0f + __expf(-x));
}

// ===========================================================================
// Phase 1: pre-gates  G[b*T+t, n] = xs[b,t,:] @ Wx[:,n] + bias[n]
// M=131072, K=256, N=512.  WG = 256 thr / 8 waves; WG tile 128(M) x 64(N).
// Double-buffered LDS tiles -> one barrier per 32-wide K chunk.
// ===========================================================================

// stage A (128x32 f32->bf16, packed b128 stores) and B (32x64, transposed to
// [n][k] with k-pair packed b32 stores)
__device__ __forceinline__ void stage_tiles(
    int tid, int kk, int R0, int C0,
    const float* __restrict__ xs, const float* __restrict__ Wx,
    __bf16 (&A)[128][40], __bf16 (&Bm)[64][40])
{
    {   // A: each thread converts 16 consecutive f32 of one row -> 2 x b128
        const int r  = tid >> 1;
        const int cs = (tid & 1) * 16;
        const float* src = xs + (size_t)(R0 + r) * OBS_ + kk + cs;
        v8bf pa, pb;
        #pragma unroll
        for (int j = 0; j < 8; ++j) pa[j] = (__bf16)src[j];
        #pragma unroll
        for (int j = 0; j < 8; ++j) pb[j] = (__bf16)src[8 + j];
        *(v8bf*)&A[r][cs]     = pa;
        *(v8bf*)&A[r][cs + 8] = pb;
    }
    {   // B: thread -> column n, 4 k-pairs; coalesced global, b32 LDS stores
        const int n  = tid & 63;
        const int kq = (tid >> 6) * 4;
        #pragma unroll
        for (int q = 0; q < 4; ++q) {
            const int k = 2 * (kq + q);
            v2bf p;
            p[0] = (__bf16)Wx[(size_t)(kk + k)     * N4_ + C0 + n];
            p[1] = (__bf16)Wx[(size_t)(kk + k + 1) * N4_ + C0 + n];
            *(v2bf*)&Bm[n][k] = p;
        }
    }
}

__device__ __forceinline__ void gemm_chunk(
    int w, int half, int l16,
    const __bf16 (&A)[128][40], const __bf16 (&Bm)[64][40], v8f (&acc)[4])
{
    // A fragment (16x32 bf16): per-lane K groups {8h..8h+7} and {16+8h..}
    v16bf a;
    {
        const v8bf lo = *(const v8bf*)&A[16 * w + l16][8 * half];
        const v8bf hi = *(const v8bf*)&A[16 * w + l16][16 + 8 * half];
        #pragma unroll
        for (int j = 0; j < 8; ++j) { a[j] = lo[j]; a[8 + j] = hi[j]; }
    }
    #pragma unroll
    for (int nt = 0; nt < 4; ++nt) {
        // B fragment (32x16 bf16): per-lane K groups {8h..8h+7} and {16+8h..}
        v16bf bf;
        const v8bf lo = *(const v8bf*)&Bm[nt * 16 + l16][8 * half];
        const v8bf hi = *(const v8bf*)&Bm[nt * 16 + l16][16 + 8 * half];
        #pragma unroll
        for (int j = 0; j < 8; ++j) { bf[j] = lo[j]; bf[8 + j] = hi[j]; }
        acc[nt] = __builtin_amdgcn_wmma_f32_16x16x32_bf16(
            false, a, false, bf, (short)0, acc[nt], false, false);
    }
}

__global__ __launch_bounds__(256)
void pregate_gemm(const float* __restrict__ xs, const float* __restrict__ Wx,
                  const float* __restrict__ bias, float* __restrict__ G)
{
    __shared__ __align__(16) __bf16 Alds[2][128][40];  // double-buffered
    __shared__ __align__(16) __bf16 Blds[2][64][40];

    const int tid  = threadIdx.x;
    const int w    = tid >> 5;
    const int l    = tid & 31;
    const int half = (l >> 4) & 1;
    const int l16  = l & 15;

    const int R0 = blockIdx.x * 128;  // M base
    const int C0 = blockIdx.y * 64;   // N base

    v8f acc[4];
    #pragma unroll
    for (int n = 0; n < 4; ++n)
        #pragma unroll
        for (int r = 0; r < 8; ++r) acc[n][r] = 0.0f;

    stage_tiles(tid, 0, R0, C0, xs, Wx, Alds[0], Blds[0]);
    __syncthreads();

    #pragma unroll
    for (int kk = 0; kk < OBS_; kk += 64) {
        // stage next chunk into buf1 while computing on buf0
        stage_tiles(tid, kk + 32, R0, C0, xs, Wx, Alds[1], Blds[1]);
        gemm_chunk(w, half, l16, Alds[0], Blds[0], acc);
        __syncthreads();

        if (kk + 64 < OBS_)
            stage_tiles(tid, kk + 64, R0, C0, xs, Wx, Alds[0], Blds[0]);
        gemm_chunk(w, half, l16, Alds[1], Blds[1], acc);
        __syncthreads();
    }

    // store D frags (+ LSTM bias folded in)
    #pragma unroll
    for (int nt = 0; nt < 4; ++nt) {
        const int col = C0 + nt * 16 + l16;
        const float bv = bias[col];
        #pragma unroll
        for (int r = 0; r < 8; ++r) {
            const int row = R0 + 16 * w + 8 * half + r;
            G[(size_t)row * N4_ + col] = acc[nt][r] + bv;
        }
    }
}

// ===========================================================================
// Phase 2: per-batch-tile LSTM recurrence + dense heads.
// 16 WGs (16 batch rows each), 256 thr / 8 waves. WhT resident bf16 in LDS.
// Wave w owns gate family {16w.., 128+16w.., 256+16w.., 384+16w..} so the
// pointwise i/f/g/o update is register-only and c never leaves registers.
// h double-buffered in LDS -> ONE barrier per timestep.
// ===========================================================================
__device__ __forceinline__ void lstm_step(
    int t, int b0, int w, int half, int l16, int jc,
    const float* __restrict__ G,
    const __bf16 (&cur)[16][H_ + 8], __bf16 (&nxt)[16][H_ + 8],
    const __bf16 (&WhT)[N4_][H_ + 8], float (&creg)[8])
{
    // pre-gate loads for this lane's (m, jc); issued before the WMMA chain
    float pre[4][8];
    #pragma unroll
    for (int r = 0; r < 8; ++r) {
        const int m = 8 * half + r;
        const float* gp = G + ((size_t)(b0 + m) * T_ + t) * N4_ + jc;
        pre[0][r] = gp[0];        // i
        pre[1][r] = gp[128];      // f
        pre[2][r] = gp[256];      // g
        pre[3][r] = gp[384];      // o
        if (t + 8 < T_)  // warm L2 ahead of the serial chain
            __builtin_prefetch(gp + 8 * N4_ + 128 * (w & 3), 0, 1);
    }

    v8f acc[4];
    #pragma unroll
    for (int n = 0; n < 4; ++n)
        #pragma unroll
        for (int r = 0; r < 8; ++r) acc[n][r] = 0.0f;

    #pragma unroll
    for (int kc = 0; kc < 4; ++kc) {  // K = 128, chunks of 32
        const int k0 = 32 * kc;
        v16bf a;
        {
            const v8bf lo = *(const v8bf*)&cur[l16][k0 + 8 * half];
            const v8bf hi = *(const v8bf*)&cur[l16][k0 + 16 + 8 * half];
            #pragma unroll
            for (int j = 0; j < 8; ++j) { a[j] = lo[j]; a[8 + j] = hi[j]; }
        }
        #pragma unroll
        for (int nt = 0; nt < 4; ++nt) {
            const int n0 = 16 * w + nt * 128;  // i/f/g/o family tile
            v16bf bf;
            const v8bf lo = *(const v8bf*)&WhT[n0 + l16][k0 + 8 * half];
            const v8bf hi = *(const v8bf*)&WhT[n0 + l16][k0 + 16 + 8 * half];
            #pragma unroll
            for (int j = 0; j < 8; ++j) { bf[j] = lo[j]; bf[8 + j] = hi[j]; }
            acc[nt] = __builtin_amdgcn_wmma_f32_16x16x32_bf16(
                false, a, false, bf, (short)0, acc[nt], false, false);
        }
    }

    // register-only LSTM pointwise update; write h_{t+1} into the OTHER buffer
    #pragma unroll
    for (int r = 0; r < 8; ++r) {
        const float iv = sigmoidf_(acc[0][r] + pre[0][r]);
        const float fv = sigmoidf_(acc[1][r] + pre[1][r]);
        const float gv = tanhf(acc[2][r] + pre[2][r]);
        const float ov = sigmoidf_(acc[3][r] + pre[3][r]);
        creg[r] = fv * creg[r] + iv * gv;
        const float hv = ov * tanhf(creg[r]);
        nxt[8 * half + r][jc] = (__bf16)hv;
    }
    __syncthreads();  // h_{t+1} visible; double-buffer => single barrier/step
}

__global__ __launch_bounds__(256)
void lstm_recurrence(const float* __restrict__ G, const float* __restrict__ Wh,
                     const float* __restrict__ W1, const float* __restrict__ b1,
                     const float* __restrict__ W2, const float* __restrict__ b2,
                     float* __restrict__ out)
{
    __shared__ __align__(16) __bf16 WhT[N4_][H_ + 8];    // 136 KiB
    __shared__ __align__(16) __bf16 hld[2][16][H_ + 8];  // double-buffered h
    __shared__ __align__(16) float  obuf[16][H_];

    const int tid  = threadIdx.x;
    const int w    = tid >> 5;
    const int l    = tid & 31;
    const int half = (l >> 4) & 1;
    const int l16  = l & 15;
    const int b0   = blockIdx.x * 16;
    const int jc   = 16 * w + l16;

    // stage Wh (128x512 f32, k-major) -> WhT[n][k] bf16, packed b128 stores
    for (int i = tid; i < N4_ * (H_ / 8); i += 256) {  // 8192 b128 slots
        const int n  = i & 511;
        const int ko = i >> 9;  // 0..15, covers k = 8*ko .. 8*ko+7
        v8bf p;
        #pragma unroll
        for (int j = 0; j < 8; ++j)
            p[j] = (__bf16)Wh[(size_t)(8 * ko + j) * N4_ + n];
        *(v8bf*)&WhT[n][8 * ko] = p;
    }
    for (int i = tid; i < 2 * 16 * (H_ + 8); i += 256)
        ((__bf16*)hld)[i] = (__bf16)0.0f;
    __syncthreads();

    float creg[8];
    #pragma unroll
    for (int r = 0; r < 8; ++r) creg[r] = 0.0f;

    for (int t = 0; t < T_; t += 2) {
        lstm_step(t,     b0, w, half, l16, jc, G, hld[0], hld[1], WhT, creg);
        lstm_step(t + 1, b0, w, half, l16, jc, G, hld[1], hld[0], WhT, creg);
    }
    // T_ even -> final h lives in hld[0]

    // ---- epilogue: out = relu(h @ W1 + b1) ----
    #pragma unroll
    for (int e = 0; e < 8; ++e) {
        const int idx = tid + 256 * e;  // 2048 outputs
        const int m = idx >> 7, j = idx & 127;
        float s = b1[j];
        for (int k = 0; k < H_; ++k)
            s += (float)hld[0][m][k] * W1[k * H_ + j];
        obuf[m][j] = s > 0.0f ? s : 0.0f;
    }
    __syncthreads();

    // ---- p = out @ W2 + b2 ; split mean / logvar into d_out ----
    #pragma unroll
    for (int e = 0; e < 2; ++e) {
        const int el = tid + 256 * e;   // 512 outputs
        const int m = el >> 5, q = el & 31;
        float s = b2[q];
        for (int k = 0; k < H_; ++k)
            s += obuf[m][k] * W2[k * 32 + q];
        const int gb = b0 + m;
        if (q < LAT_) out[gb * LAT_ + q] = s;                       // mean
        else          out[B_ * LAT_ + gb * LAT_ + (q - LAT_)] = s;  // logvar
    }
}

// ===========================================================================
extern "C" void kernel_launch(void* const* d_in, const int* in_sizes, int n_in,
                              void* d_out, int out_size, void* d_ws, size_t ws_size,
                              hipStream_t stream)
{
    (void)in_sizes; (void)n_in; (void)out_size; (void)ws_size;
    const float* xs = (const float*)d_in[0];
    const float* Wx = (const float*)d_in[1];
    const float* Wh = (const float*)d_in[2];
    const float* bb = (const float*)d_in[3];
    const float* W1 = (const float*)d_in[4];
    const float* b1 = (const float*)d_in[5];
    const float* W2 = (const float*)d_in[6];
    const float* b2 = (const float*)d_in[7];

    float* G = (float*)d_ws;  // [B*T, 4H] f32 pre-gates, 256 MiB scratch

    dim3 g1((B_ * T_) / 128, N4_ / 64);  // 1024 x 8
    pregate_gemm<<<g1, 256, 0, stream>>>(xs, Wx, bb, G);
    lstm_recurrence<<<B_ / 16, 256, 0, stream>>>(G, Wh, W1, b1, W2, b2,
                                                 (float*)d_out);
}
